// CoarseMatching_37512244364108
// MI455X (gfx1250) — matrix operand order repaired and checked by reading, not compiled, not run on hardware
//
#include <hip/hip_runtime.h>
#include <hip/hip_bf16.h>

typedef __attribute__((ext_vector_type(16))) _Float16 v16h;
typedef __attribute__((ext_vector_type(8)))  float    v8f;

// Problem constants (match reference setup_inputs)
constexpr int N = 4, L = 2048, C = 256;
constexpr int H1C = 60, W1C = 80, S = H1C * W1C;   // 4800
constexpr int KB = C / 32;                          // 8 k-steps of 32
constexpr float TEMPERATURE = 0.08f;
constexpr float THR = 0.2f;
constexpr int   BORDER = 2;
constexpr float NEG_INF_V = -1.0e9f;
constexpr float QSCALE = 8.0f;                      // 480/60
constexpr float FLT_NEG = -3.402823466e38f;

// Online-softmax accumulator combine: (m,d) <- (m,d) (+) (m2,d2)
__device__ __forceinline__ void sm_combine(float& m, float& d, float m2, float d2) {
    float mn = fmaxf(m, m2);
    d = d * expf(m - mn) + d2 * expf(m2 - mn);
    m = mn;
}

// ---------------------------------------------------------------------------
// Pack A (feat_db_3d) fp32 -> f16 in WMMA A-fragment lane order.
// A 16x32 as packed here (contiguous k, same order as B): lane = half*16 + m; element e:
//   k = half*16 + e
// Fragment storage: [n][tm][kb][lane][16] contiguous halves.
// ---------------------------------------------------------------------------
__global__ __launch_bounds__(256)
void pack_a_kernel(const float* __restrict__ f3d, _Float16* __restrict__ Ap) {
    int t = blockIdx.x * blockDim.x + threadIdx.x;
    const int total = N * (L / 16) * KB * 32;
    if (t >= total) return;
    int lane = t & 31; int rest = t >> 5;
    int kb = rest % KB; rest /= KB;
    int tm = rest % (L / 16); int n = rest / (L / 16);
    int m = lane & 15, half = lane >> 4;
    int l = tm * 16 + m;
    const float* src = f3d + ((size_t)n * L + l) * C + kb * 32;
    v16h frag;
#pragma unroll
    for (int e = 0; e < 16; ++e) {
        int vg = e >> 1;
        int k = half * 16 + e;
        frag[e] = (_Float16)src[k];
    }
    *reinterpret_cast<v16h*>(Ap + (size_t)t * 16) = frag;
}

// ---------------------------------------------------------------------------
// Pack B (feat_query) fp32 -> f16 in WMMA B-fragment lane order.
// B 32x16 as packed here (contiguous k, same order as A): lane = half*16 + nCol; per lane holds k = half*16 + e.
// feat_query is [S][C] row-major -> the 16 k's are contiguous in memory.
// ---------------------------------------------------------------------------
__global__ __launch_bounds__(256)
void pack_b_kernel(const float* __restrict__ fq, _Float16* __restrict__ Bp) {
    int t = blockIdx.x * blockDim.x + threadIdx.x;
    const int total = N * (S / 16) * KB * 32;
    if (t >= total) return;
    int lane = t & 31; int rest = t >> 5;
    int kb = rest % KB; rest /= KB;
    int ts = rest % (S / 16); int n = rest / (S / 16);
    int col = ts * 16 + (lane & 15);
    int half = lane >> 4;
    const float* src = fq + ((size_t)n * S + col) * C + kb * 32 + half * 16;
    v16h frag;
#pragma unroll
    for (int e = 0; e < 16; ++e) frag[e] = (_Float16)src[e];
    *reinterpret_cast<v16h*>(Bp + (size_t)t * 16) = frag;
}

// ---------------------------------------------------------------------------
// WMMA GEMM: sim[n][l][s] = scale * dot(f3d[l,:], fq[s,:]) (+ mask term)
// One wave32 per block; wave owns 4x2 tiles of 16x16 (64 rows x 32 cols).
// A (4 MB) + B (9.8 MB) panels fit in the 192 MB L2, so the heavy re-reads
// across the 32x150 block grid are L2 hits; HBM cost is the 157 MB sim write.
// ---------------------------------------------------------------------------
__global__ __launch_bounds__(32)
void gemm_sim_kernel(const _Float16* __restrict__ Ap, const _Float16* __restrict__ Bp,
                     const unsigned char* __restrict__ maskq, float* __restrict__ sim) {
    const int lane = threadIdx.x;
    const int half = lane >> 4;
    const int nIdx = lane & 15;
    const int mb = blockIdx.x, nb = blockIdx.y, n = blockIdx.z;
    const int tmBase = mb * 4;
    const int tnBase = nb * 2;

    v8f acc[4][2] = {};

    const _Float16* aB = Ap + (size_t)n * (L / 16) * KB * 512;
    const _Float16* bB = Bp + (size_t)n * (S / 16) * KB * 512;

    for (int kb = 0; kb < KB; ++kb) {
        v16h a[4], b[2];
#pragma unroll
        for (int i = 0; i < 4; ++i)
            a[i] = *reinterpret_cast<const v16h*>(
                aB + (((size_t)(tmBase + i) * KB + kb) * 512) + lane * 16);
#pragma unroll
        for (int j = 0; j < 2; ++j)
            b[j] = *reinterpret_cast<const v16h*>(
                bB + (((size_t)(tnBase + j) * KB + kb) * 512) + lane * 16);
#pragma unroll
        for (int i = 0; i < 4; ++i)
#pragma unroll
            for (int j = 0; j < 2; ++j)
                acc[i][j] = __builtin_amdgcn_wmma_f32_16x16x32_f16(
                    false, a[i], false, b[j], (short)0, acc[i][j], false, false);
    }

    const float scale = (1.0f / (float)C) * (1.0f / (TEMPERATURE + 1e-4f));
#pragma unroll
    for (int i = 0; i < 4; ++i) {
#pragma unroll
        for (int j = 0; j < 2; ++j) {
            int s = (tnBase + j) * 16 + nIdx;
            float madd = maskq[(size_t)n * S + s] ? 0.0f : NEG_INF_V;
#pragma unroll
            for (int r = 0; r < 8; ++r) {
                int l = (tmBase + i) * 16 + half * 8 + r;
                sim[((size_t)n * L + l) * (size_t)S + s] = acc[i][j][r] * scale + madd;
            }
        }
    }
}

// ---------------------------------------------------------------------------
// Row (axis=2, over S) softmax stats, single streaming pass (online softmax).
// ---------------------------------------------------------------------------
__global__ __launch_bounds__(256)
void row_stats_kernel(const float* __restrict__ sim,
                      float* __restrict__ rmax, float* __restrict__ rsum) {
    __shared__ float redm[256];
    __shared__ float redd[256];
    const int row = blockIdx.x;                 // n*L + l
    const float* p = sim + (size_t)row * S;
    const int tid = threadIdx.x;
    float m = FLT_NEG, d = 0.0f;
    for (int s = tid; s < S; s += 256) sm_combine(m, d, p[s], 1.0f);
    redm[tid] = m; redd[tid] = d; __syncthreads();
    for (int off = 128; off > 0; off >>= 1) {
        if (tid < off) {
            float m2 = redm[tid + off], d2 = redd[tid + off];
            float mm = redm[tid], dd = redd[tid];
            sm_combine(mm, dd, m2, d2);
            redm[tid] = mm; redd[tid] = dd;
        }
        __syncthreads();
    }
    if (tid == 0) { rmax[row] = redm[0]; rsum[row] = redd[0]; }
}

// ---------------------------------------------------------------------------
// Column (axis=1, over L) softmax stats: thread-per-column (coalesced),
// single streaming pass (online softmax) -> halves the HBM reads vs two-pass
// since the full 157 MB tensor cannot sit in L2 between passes.
// ---------------------------------------------------------------------------
__global__ __launch_bounds__(256)
void col_stats_kernel(const float* __restrict__ sim,
                      float* __restrict__ cmax, float* __restrict__ csum) {
    int s = blockIdx.x * blockDim.x + threadIdx.x;
    int n = blockIdx.y;
    if (s >= S) return;
    const float* p = sim + (size_t)n * L * S + s;
    float m = FLT_NEG, d = 0.0f;
    for (int l = 0; l < L; ++l) sm_combine(m, d, p[(size_t)l * S], 1.0f);
    cmax[n * S + s] = m; csum[n * S + s] = d;
}

// ---------------------------------------------------------------------------
// Fused: conf = softmax_axis1 * softmax_axis2 (in place) AND row-wise max of
// conf for the mutual-NN test. Block-per-row keeps conf in registers so the
// row max is free, eliminating a full 157 MB re-read pass.
// ---------------------------------------------------------------------------
__global__ __launch_bounds__(256)
void conf_rowmax_kernel(float* __restrict__ simconf,
                        const float* __restrict__ rmax, const float* __restrict__ rsum,
                        const float* __restrict__ cmax, const float* __restrict__ csum,
                        float* __restrict__ rowmaxc) {
    __shared__ float red[256];
    const int row = blockIdx.x;                 // n*L + l
    const int n = row / L;
    const int tid = threadIdx.x;
    float* p = simconf + (size_t)row * S;
    const float rm = rmax[row];
    const float rinv = 1.0f / rsum[row];
    const float* cm = cmax + (size_t)n * S;
    const float* cs = csum + (size_t)n * S;
    float best = FLT_NEG;
    for (int s = tid; s < S; s += 256) {
        float v = p[s];
        float c = (expf(v - rm) * rinv) * (expf(v - cm[s]) / cs[s]);
        p[s] = c;
        best = fmaxf(best, c);
    }
    red[tid] = best; __syncthreads();
    for (int off = 128; off > 0; off >>= 1) {
        if (tid < off) red[tid] = fmaxf(red[tid], red[tid + off]);
        __syncthreads();
    }
    if (tid == 0) rowmaxc[row] = red[0];
}

// Column-wise max of conf (mutual-NN, axis=1)
__global__ __launch_bounds__(256)
void col_max_kernel(const float* __restrict__ conf, float* __restrict__ colmaxc) {
    int s = blockIdx.x * blockDim.x + threadIdx.x;
    int n = blockIdx.y;
    if (s >= S) return;
    const float* p = conf + (size_t)n * L * S + s;
    float m = FLT_NEG;
    for (int l = 0; l < L; ++l) m = fmaxf(m, p[(size_t)l * S]);
    colmaxc[n * S + s] = m;
}

// ---------------------------------------------------------------------------
// Match extraction per (n,l): threshold + border + mutual-NN, first-True idx.
// ---------------------------------------------------------------------------
__global__ __launch_bounds__(256)
void match_kernel(const float* __restrict__ conf,
                  const float* __restrict__ rowmaxc, const float* __restrict__ colmaxc,
                  float* __restrict__ out) {
    __shared__ int redi[256];
    const int row = blockIdx.x;                 // n*L + l
    const int n = row / L;
    const int tid = threadIdx.x;
    const float* p = conf + (size_t)row * S;
    const float rm = rowmaxc[row];
    int best = S;
    for (int s = tid; s < S; s += 256) {
        if (s + 1024 < S) __builtin_prefetch(&p[s + 1024], 0, 1);
        float c = p[s];
        int hh = s / W1C, ww = s % W1C;
        bool flag = (c > THR) && (hh >= BORDER) && (ww >= BORDER) &&
                    (c == rm) && (c == colmaxc[(size_t)n * S + s]);
        if (flag && s < best) best = s;
    }
    redi[tid] = best; __syncthreads();
    for (int off = 128; off > 0; off >>= 1) {
        if (tid < off) redi[tid] = min(redi[tid], redi[tid + off]);
        __syncthreads();
    }
    if (tid == 0) {
        int j = redi[0];
        bool found = (j < S);
        if (!found) j = 0;                      // argmax of all-False == 0
        float mc = found ? p[j] : 0.0f;
        const size_t NLS = (size_t)N * L * S;
        const size_t NL  = (size_t)N * L;
        out[NLS + row]              = found ? 1.0f : 0.0f;   // mask_v
        out[NLS + NL + row]         = (float)j;              // all_j_ids
        out[NLS + 2 * NL + row]     = mc;                    // mconf
        out[NLS + 3 * NL + 2 * (size_t)row + 0] = (float)(j % W1C) * QSCALE;
        out[NLS + 3 * NL + 2 * (size_t)row + 1] = (float)(j / W1C) * QSCALE;
    }
}

// keypoints3d passthrough
__global__ __launch_bounds__(256)
void copy_kpts_kernel(const float* __restrict__ kpts, float* __restrict__ out) {
    size_t i = (size_t)blockIdx.x * blockDim.x + threadIdx.x;
    const size_t total = (size_t)N * L * 3;
    if (i < total)
        out[(size_t)N * L * S + 5 * (size_t)N * L + i] = kpts[i];
}

extern "C" void kernel_launch(void* const* d_in, const int* in_sizes, int n_in,
                              void* d_out, int out_size, void* d_ws, size_t ws_size,
                              hipStream_t stream) {
    const float* f3d  = (const float*)d_in[0];
    const float* fq   = (const float*)d_in[1];
    const float* kpts = (const float*)d_in[2];
    const unsigned char* maskq = (const unsigned char*)d_in[3];
    float* out = (float*)d_out;

    // Workspace carve-out (256B aligned slices)
    char* ws = (char*)d_ws;
    size_t off = 0;
    auto carve = [&](size_t bytes) -> char* {
        char* p = ws + off;
        off = (off + bytes + 255) & ~(size_t)255;
        return p;
    };
    _Float16* Ap    = (_Float16*)carve((size_t)N * L * C * sizeof(_Float16));
    _Float16* Bp    = (_Float16*)carve((size_t)N * S * C * sizeof(_Float16));
    float* rmax     = (float*)carve((size_t)N * L * sizeof(float));
    float* rsum     = (float*)carve((size_t)N * L * sizeof(float));
    float* cmax     = (float*)carve((size_t)N * S * sizeof(float));
    float* csum     = (float*)carve((size_t)N * S * sizeof(float));
    float* rowmaxc  = (float*)carve((size_t)N * L * sizeof(float));
    float* colmaxc  = (float*)carve((size_t)N * S * sizeof(float));

    // 1) pack fp32 -> f16 WMMA fragment layout
    {
        int total = N * (L / 16) * KB * 32;
        pack_a_kernel<<<(total + 255) / 256, 256, 0, stream>>>(f3d, Ap);
    }
    {
        int total = N * (S / 16) * KB * 32;
        pack_b_kernel<<<(total + 255) / 256, 256, 0, stream>>>(fq, Bp);
    }

    // 2) WMMA GEMM -> sim written into the conf region of d_out
    gemm_sim_kernel<<<dim3(L / 64, S / 32, N), 32, 0, stream>>>(Ap, Bp, maskq, out);

    // 3) dual softmax stats (single streaming pass each)
    row_stats_kernel<<<N * L, 256, 0, stream>>>(out, rmax, rsum);
    col_stats_kernel<<<dim3((S + 255) / 256, N), 256, 0, stream>>>(out, cmax, csum);

    // 4) in-place conf normalize fused with mutual-NN row max
    conf_rowmax_kernel<<<N * L, 256, 0, stream>>>(out, rmax, rsum, cmax, csum, rowmaxc);

    // 5) mutual-NN column max over conf
    col_max_kernel<<<dim3((S + 255) / 256, N), 256, 0, stream>>>(out, colmaxc);

    // 6) match extraction + passthrough outputs
    match_kernel<<<N * L, 256, 0, stream>>>(out, rowmaxc, colmaxc, out);
    copy_kpts_kernel<<<((N * L * 3) + 255) / 256, 256, 0, stream>>>(kpts, out);
}
